// Position_encode_71880572666237
// MI455X (gfx1250) — matrix operand order repaired and checked by reading, not compiled, not run on hardware
//
#include <hip/hip_runtime.h>
#include <math.h>

#define N_NODES 100000
#define DIM     64
#define BSEL    8192
#define KP      32
#define KN      100

typedef __attribute__((ext_vector_type(16))) _Float16       v16h;
typedef __attribute__((ext_vector_type(16))) unsigned short v16u;
typedef __attribute__((ext_vector_type(8)))  float          v8f;
typedef __attribute__((ext_vector_type(4)))  float          v4f;
typedef __attribute__((ext_vector_type(2)))  _Float16       v2h;

// 32-byte fragment viewed as: WMMA f16 operand / ushort lanes (for abs) / two 16B load chunks
union AF {
    v16h v;
    v16u u;
    v4f  f4[2];
};

__device__ __forceinline__ void atomic_add_f32(float* p, float v) {
    __hip_atomic_fetch_add(p, v, __ATOMIC_RELAXED, __HIP_MEMORY_SCOPE_AGENT);
}

// ---------------------------------------------------------------------------
// Workspace layout:
//   ws + 0   : float acc[3]   (deg SSE, sum t_adj, sum t_deg)
//   ws + 64  : _Float16 Wh[64] (f16 copy of W_d, built once)
//   ws + 256 : _Float16 Z[N*D] (fp16 sigmoid(P), 12.8 MB)
__global__ void k_init(float* __restrict__ acc, const float* __restrict__ Wd,
                       _Float16* __restrict__ Wh) {
    int t = threadIdx.x;
    if (t < 3) acc[t] = 0.0f;
    if (t < DIM) Wh[t] = (_Float16)Wd[t];
}

// ---------------------------------------------------------------------------
// Z = sigmoid(P), stored fp16 (halves the dominant L2 gather traffic)
__global__ void k_sigmoid(const float* __restrict__ P, _Float16* __restrict__ Z) {
    int i = blockIdx.x * blockDim.x + threadIdx.x;
    const int total2 = N_NODES * DIM / 2;
    if (i < total2) {
        float p0 = P[2 * i + 0];
        float p1 = P[2 * i + 1];
        float z0 = 1.0f / (1.0f + __expf(-p0));
        float z1 = 1.0f / (1.0f + __expf(-p1));
        v2h h;
        h[0] = (_Float16)z0;
        h[1] = (_Float16)z1;
        ((v2h*)Z)[i] = h;
    }
}

// ---------------------------------------------------------------------------
// Degree loss: one wave handles a 16-node tile; y = Z_tile @ W via WMMA with
// W broadcast across all 16 columns of B. Accumulates sum (y - deg)^2.
__global__ void k_degree(const _Float16* __restrict__ Z,
                         const _Float16* __restrict__ Wh,
                         const float* __restrict__ deg,
                         float* __restrict__ acc) {
    const int TILES = N_NODES / 16;                 // 6250, exact
    int wave = (blockIdx.x * blockDim.x + threadIdx.x) >> 5;
    int lane = threadIdx.x & 31;
    if (wave >= TILES) return;                      // wave-uniform: EXEC stays all-1s

    int hi = (lane >> 4) & 1;                       // lane half
    int m  = lane & 15;                             // A-row owned by this lane

    // A rows (ISA 16-bit A 16x32 layout): hi=0 -> dim chunks {0,2} / {4,6}; hi=1 -> {1,3} / {5,7}
    const v4f* row = (const v4f*)(Z + (size_t)(wave * 16 + m) * DIM);
    AF a0, a1;
    a0.f4[0] = row[0 + hi]; a0.f4[1] = row[2 + hi];
    a1.f4[0] = row[4 + hi]; a1.f4[1] = row[6 + hi];

    // B (32x16, columns all equal = W): same chunk pattern as A: hi=0 -> chunks {0,2} / {4,6}; hi=1 -> {1,3} / {5,7}
    // Wh is pre-converted f16: two b128 loads per step.
    const v4f* wh4 = (const v4f*)Wh;                // 8 halfs per chunk
    AF w0, w1;
    w0.f4[0] = wh4[0 + hi]; w0.f4[1] = wh4[2 + hi];
    w1.f4[0] = wh4[4 + hi]; w1.f4[1] = wh4[6 + hi];

    v8f c = {};
    c = __builtin_amdgcn_wmma_f32_16x16x32_f16(false, a0.v, false, w0.v, (short)0, c, false, false);
    c = __builtin_amdgcn_wmma_f32_16x16x32_f16(false, a1.v, false, w1.v, (short)0, c, false, false);

    // D layout: vgpr r = row r (lanes 0-15) / row r+8 (lanes 16-31); all columns identical.
    float s = 0.0f;
    for (int r = 0; r < 8; ++r) {
        int node = wave * 16 + hi * 8 + r;
        float d  = c[r] - deg[node];
        s += d * d;
    }
    s += __shfl_xor(s, 16, 32);                     // rows 0-7 + rows 8-15
    if (lane == 0) atomic_add_f32(&acc[0], s);
}

// ---------------------------------------------------------------------------
// Contrast: one wave per (selected node b, contrast type). 16 gathered rows per
// WMMA group; ham row-sums via WMMA against a ones B-matrix; packed f16 abs-diff
// on VALU; exp/log-sum-exp in f32.
__global__ void k_contrast(const _Float16* __restrict__ Z,
                           const int* __restrict__ sel,
                           const int* __restrict__ pos_adj,
                           const int* __restrict__ neg_adj,
                           const int* __restrict__ pos_deg,
                           const int* __restrict__ neg_deg,
                           float* __restrict__ acc) {
    int wave = (blockIdx.x * blockDim.x + threadIdx.x) >> 5;   // [0, 2*BSEL)
    int lane = threadIdx.x & 31;
    int b  = wave & (BSEL - 1);
    int ct = wave >> 13;                                       // 0: adj, 1: deg-dist
    const int* __restrict__ pidx = ct ? pos_deg : pos_adj;
    const int* __restrict__ nidx = ct ? neg_deg : neg_adj;

    int hi = (lane >> 4) & 1;
    int m  = lane & 15;

    // Z[sel[b]] fragments (same chunk pattern as A rows)
    const v4f* zi_row = (const v4f*)(Z + (size_t)sel[b] * DIM);
    AF zi0, zi1;
    zi0.f4[0] = zi_row[0 + hi]; zi0.f4[1] = zi_row[2 + hi];
    zi1.f4[0] = zi_row[4 + hi]; zi1.f4[1] = zi_row[6 + hi];

    v16h ones;
    for (int i = 0; i < 16; ++i) ones[i] = (_Float16)1.0f;
    v16u amask;
    for (int i = 0; i < 16; ++i) amask[i] = 0x7FFFu;           // f16 abs = clear sign

    float sum_pos = 0.0f, sum_exp = 0.0f;
    const int NGROUPS = 2 + (KN + 15) / 16;                    // 2 pos + 7 neg groups

    for (int g = 0; g < NGROUPS; ++g) {
        int pr = g * 16 + m;                                   // pair-row index
        int j;
        if (g < 2) {                                           // wave-uniform branch
            j = pidx[(size_t)b * KP + pr];
        } else {
            int kn = pr - 32;
            if (kn > KN - 1) kn = KN - 1;                      // clamp ragged tail
            j = nidx[(size_t)b * KN + kn];
        }
        const v4f* row = (const v4f*)(Z + (size_t)j * DIM);
        AF a0, a1;
        a0.f4[0] = row[0 + hi]; a0.f4[1] = row[2 + hi];
        a1.f4[0] = row[4 + hi]; a1.f4[1] = row[6 + hi];

        a0.v = a0.v - zi0.v;  a0.u = a0.u & amask;             // packed |zi - zj|
        a1.v = a1.v - zi1.v;  a1.u = a1.u & amask;

        v8f c = {};
        c = __builtin_amdgcn_wmma_f32_16x16x32_f16(false, a0.v, false, ones, (short)0, c, false, false);
        c = __builtin_amdgcn_wmma_f32_16x16x32_f16(false, a1.v, false, ones, (short)0, c, false, false);

        if (g < 2) {                                           // positives: plain sum
            float s = c[0] + c[1] + c[2] + c[3] + c[4] + c[5] + c[6] + c[7];
            s += __shfl_xor(s, 16, 32);
            sum_pos += s;
        } else {                                               // negatives: sum exp(ham)
            float es = 0.0f;
            for (int r = 0; r < 8; ++r) {
                int kn = g * 16 + hi * 8 + r - 32;
                float e = (kn < KN) ? __expf(c[r]) : 0.0f;     // mask padded rows
                es += e;
            }
            es += __shfl_xor(es, 16, 32);
            sum_exp += es;
        }
    }

    if (lane == 0) {
        float t = -sum_pos * (1.0f / (float)KP) + __logf(sum_exp);
        atomic_add_f32(&acc[1 + ct], t);
    }
}

// ---------------------------------------------------------------------------
__global__ void k_finalize(const float* __restrict__ acc, float* __restrict__ out) {
    if (threadIdx.x == 0) {
        out[0] = sqrtf(acc[0]);
        out[1] = acc[1] * (1.0f / (float)N_NODES);
        out[2] = acc[2] * (1.0f / (float)N_NODES);
    }
}

// ---------------------------------------------------------------------------
extern "C" void kernel_launch(void* const* d_in, const int* in_sizes, int n_in,
                              void* d_out, int out_size, void* d_ws, size_t ws_size,
                              hipStream_t stream) {
    const float* P       = (const float*)d_in[0];
    const float* Wd      = (const float*)d_in[1];
    const float* deg     = (const float*)d_in[2];
    const int*   sel     = (const int*)d_in[3];
    const int*   pos_adj = (const int*)d_in[4];
    const int*   neg_adj = (const int*)d_in[5];
    const int*   pos_deg = (const int*)d_in[6];
    const int*   neg_deg = (const int*)d_in[7];

    float*    acc = (float*)d_ws;                        // 3 accumulators
    _Float16* Wh  = (_Float16*)((char*)d_ws + 64);       // f16 W_d (64 halfs)
    _Float16* Z   = (_Float16*)((char*)d_ws + 256);      // fp16 Z, 12.8 MB

    k_init<<<1, 64, 0, stream>>>(acc, Wd, Wh);

    const int total2 = N_NODES * DIM / 2;
    k_sigmoid<<<(total2 + 255) / 256, 256, 0, stream>>>(P, Z);

    const int deg_waves = N_NODES / 16;                  // 6250 waves, 8 waves/block
    k_degree<<<(deg_waves + 7) / 8, 256, 0, stream>>>(Z, Wh, deg, acc);

    const int ctr_blocks = (2 * BSEL) / 8;               // 16384 waves, 8 waves/block
    k_contrast<<<ctr_blocks, 256, 0, stream>>>(Z, sel, pos_adj, neg_adj, pos_deg, neg_deg, acc);

    k_finalize<<<1, 64, 0, stream>>>(acc, (float*)d_out);
}
